// BiLSTM_CRF_68822555951104
// MI455X (gfx1250) — matrix operand order repaired and checked by reading, not compiled, not run on hardware
//
#include <hip/hip_runtime.h>
#include <hip/hip_bf16.h>

// BiLSTM-CRF forward for MI455X (gfx1250), WMMA-based.
// Pipeline: f32->f16 weight/act conversion -> embedding gather ->
// WMMA GEMM xw = x@Wih^T + b (async-to-LDS double-buffered pipeline) ->
// WMMA LSTM scan (h in LDS f16, c in regs) -> classifier -> CRF NLL.

#define VOCAB 30000
#define EMB   300
#define EMBP  320          // EMB padded to multiple of 32 for WMMA K-loop
#define HID   256
#define NTAGS 9
#define BBAT  64
#define TSEQ  512
#define BT    (BBAT * TSEQ)
#define GATES 1024         // 4*HID

typedef _Float16 h8   __attribute__((ext_vector_type(8)));
typedef _Float16 v16h __attribute__((ext_vector_type(16)));
typedef float    v8f  __attribute__((ext_vector_type(8)));
typedef int      v4i  __attribute__((__vector_size__(16)));  // b128 payload

union F16x16 { v16h v; h8 h[2]; };

// ---- CDNA5 async global->LDS copy (ASYNCcnt) with portable fallback --------
#if defined(__HIP_DEVICE_COMPILE__) && \
    __has_builtin(__builtin_amdgcn_global_load_async_to_lds_b128)
#define HAVE_ASYNC_LDS 1
#else
#define HAVE_ASYNC_LDS 0
#endif

#if HAVE_ASYNC_LDS
typedef __attribute__((address_space(1))) v4i* g128p;  // global b128 ptr
typedef __attribute__((address_space(3))) v4i* l128p;  // LDS b128 ptr
#endif

// Copies 16 bytes from global to LDS. Async path issues
// global_load_async_to_lds_b128 (per-lane 16B, tracked by ASYNCcnt).
__device__ inline void copy16_g2l(const _Float16* g, _Float16* l) {
#if HAVE_ASYNC_LDS
  __builtin_amdgcn_global_load_async_to_lds_b128(
      (g128p)(uintptr_t)g, (l128p)(unsigned)(uintptr_t)l, 0, 0);
#else
  *(h8*)l = *(const h8*)g;   // register-staged fallback
#endif
}

template <int N>
__device__ inline void async_wait() {
#if HAVE_ASYNC_LDS
#if __has_builtin(__builtin_amdgcn_s_wait_asynccnt)
  __builtin_amdgcn_s_wait_asynccnt(N);
#else
  asm volatile("s_wait_asynccnt %0" ::"i"(N) : "memory");
#endif
#endif
}

// ---- WMMA fragment helpers (wave32, 16x16x32 f16 -> f32) -------------------
// A matrix 16x32 (MxK), row-major source with leading dim lda (halves).
// Lane L: M = L&15; elems 0..7 -> K = k0 + (L>>4)*8 + 0..7 ; elems 8..15 -> +16.
__device__ inline v16h load_a_frag(const _Float16* __restrict__ A, int lda,
                                   int m0, int k0, int lane) {
  F16x16 u;
  const _Float16* base = A + (size_t)(m0 + (lane & 15)) * lda + k0 + ((lane >> 4) << 3);
  u.h[0] = *(const h8*)(base);
  u.h[1] = *(const h8*)(base + 16);
  return u.v;
}

// B matrix 32x16 (KxN) supplied as W row-major (N x K): B[k][n] = W[n][k].
// Lane L: N = L&15; elems 0..7 -> K = k0 + (L>>4)*8 + 0..7 ; elems 8..15 -> +16.
__device__ inline v16h load_b_frag(const _Float16* __restrict__ W, int ldw,
                                   int n0, int k0, int lane) {
  F16x16 u;
  const _Float16* base = W + (size_t)(n0 + (lane & 15)) * ldw + k0 + ((lane >> 4) << 3);
  u.h[0] = *(const h8*)(base);
  u.h[1] = *(const h8*)(base + 16);
  return u.v;
}

__device__ inline v8f wmma_f16(v16h a, v16h b, v8f c) {
  return __builtin_amdgcn_wmma_f32_16x16x32_f16(false, a, false, b, (short)0, c,
                                                false, false);
}

__device__ inline float sigmoidf_(float x) { return 1.f / (1.f + __expf(-x)); }

// ---- Kernel: f32 -> f16 convert with optional K padding --------------------
__global__ void convert_pad_kernel(const float* __restrict__ src,
                                   _Float16* __restrict__ dst,
                                   int rows, int ksrc, int kdst) {
  size_t idx = (size_t)blockIdx.x * blockDim.x + threadIdx.x;
  size_t total = (size_t)rows * kdst;
  if (idx >= total) return;
  int k = (int)(idx % kdst);
  size_t r = idx / kdst;
  dst[idx] = (_Float16)((k < ksrc) ? src[r * ksrc + k] : 0.f);
}

// ---- Kernel: embedding gather directly to f16 (EMB padded to EMBP) ---------
__global__ void embed_kernel(const int* __restrict__ ids,
                             const float* __restrict__ emb,
                             _Float16* __restrict__ xf) {
  int token = blockIdx.x;       // 0..BT-1
  int e = threadIdx.x;          // 0..EMBP-1
  int id = ids[token];
  float v = (e < EMB) ? emb[(size_t)id * EMB + e] : 0.f;
  xf[(size_t)token * EMBP + e] = (_Float16)v;
}

// ---- Kernel: Y[M,1024] = X[M,K](f16) @ W^T[K,1024](f16) + bias, f32 out ----
// Block tile 128(M) x 64(N). Double-buffered async global->LDS staging:
// prefetch K-chunk k+1 (A: 8KB, B: 4KB) while WMMAs consume chunk k from LDS.
// 8 waves: wave = 2(M-quad of 64 rows) x 4(N tiles of 16).
__global__ __launch_bounds__(256) void gemm_bias_kernel(
    const _Float16* __restrict__ X, int lda,
    const _Float16* __restrict__ W, int ldw,
    const float* __restrict__ bias,
    float* __restrict__ Y, int ldy, int K) {
  __shared__ __align__(16) _Float16 Abuf[2][128 * 32];  // 2 x 8 KB
  __shared__ __align__(16) _Float16 Bbuf[2][64 * 32];   // 2 x 4 KB

  const int tid  = threadIdx.x;
  const int lane = tid & 31;
  const int wave = tid >> 5;                 // 0..7
  const int M0 = blockIdx.x * 128;
  const int N0 = blockIdx.y * 64;
  const int mq = (wave >> 2) * 64;           // wave M offset in tile
  const int nq = (wave & 3) * 16;            // wave N offset in tile

  // cooperative copy indices: each thread moves 16B per buffer-row group
  const int crow = tid >> 2;                 // 0..63
  const int ccol = (tid & 3) * 8;            // 0,8,16,24 (halves)

  v8f acc[4];
  v8f zero = {0.f, 0.f, 0.f, 0.f, 0.f, 0.f, 0.f, 0.f};
#pragma unroll
  for (int mt = 0; mt < 4; ++mt) acc[mt] = zero;

  const int nch = K / 32;

  // stage chunk 0 into buffer 0
  copy16_g2l(X + (size_t)(M0 + crow) * lda + ccol,      &Abuf[0][crow * 32 + ccol]);
  copy16_g2l(X + (size_t)(M0 + 64 + crow) * lda + ccol, &Abuf[0][(64 + crow) * 32 + ccol]);
  copy16_g2l(W + (size_t)(N0 + crow) * ldw + ccol,      &Bbuf[0][crow * 32 + ccol]);

  for (int kc = 0; kc < nch; ++kc) {
    if (kc + 1 < nch) {
      const int k0 = (kc + 1) * 32;
      const int nb = (kc + 1) & 1;
      copy16_g2l(X + (size_t)(M0 + crow) * lda + k0 + ccol,
                 &Abuf[nb][crow * 32 + ccol]);
      copy16_g2l(X + (size_t)(M0 + 64 + crow) * lda + k0 + ccol,
                 &Abuf[nb][(64 + crow) * 32 + ccol]);
      copy16_g2l(W + (size_t)(N0 + crow) * ldw + k0 + ccol,
                 &Bbuf[nb][crow * 32 + ccol]);
      async_wait<3>();   // chunk kc landed; chunk kc+1 (3 copies) in flight
    } else {
      async_wait<0>();   // drain
    }
    __syncthreads();

    const int cb = kc & 1;
    v16h bf = load_b_frag(&Bbuf[cb][0], 32, nq, 0, lane);
#pragma unroll
    for (int mt = 0; mt < 4; ++mt) {
      v16h af = load_a_frag(&Abuf[cb][0], 32, mq + mt * 16, 0, lane);
      acc[mt] = wmma_f16(af, bf, acc[mt]);
    }
    __syncthreads();     // protect buffer reuse (next prefetch writes cb)
  }

  const int col = N0 + nq + (lane & 15);
  const float bv = bias[col];
#pragma unroll
  for (int mt = 0; mt < 4; ++mt) {
    int rbase = M0 + mq + mt * 16 + ((lane >> 4) << 3);
#pragma unroll
    for (int r = 0; r < 8; ++r)
      Y[(size_t)(rbase + r) * ldy + col] = acc[mt][r] + bv;
  }
}

// ---- Kernel: LSTM sequential scan (one block per direction) ----------------
// blockDim = 1024 (32 wave32s). h kept in LDS as f16 (64x256 = 32KB); c kept
// in registers: each thread permanently owns 16 (batch,hid) cells.
// Per step: gates(64x1024) = h @ Whh^T via WMMA + xw[t]; elementwise update.
__global__ __launch_bounds__(1024) void lstm_scan_kernel(
    const float* __restrict__ xw,      // [2][BT][GATES], b-major rows (b*T+t)
    const _Float16* __restrict__ whh,  // [2][GATES][HID] f16
    float* __restrict__ out) {         // [BT][2*HID]
  const int dir = blockIdx.x;
  const float* xw_d = xw + (size_t)dir * BT * GATES;
  const _Float16* whh_d = whh + (size_t)dir * GATES * HID;

  const int lane = threadIdx.x & 31;
  const int wave = threadIdx.x >> 5;   // 0..31
  const int hidTile = wave >> 1;       // 0..15  -> hid block of 16
  const int mPair = wave & 1;          // 0..1   -> batch tiles {0,1} or {2,3}
  const int hid = hidTile * 16 + (lane & 15);

  __shared__ __align__(16) _Float16 hbuf[BBAT * HID];  // 32 KB

  for (int i = threadIdx.x; i < BBAT * HID; i += blockDim.x)
    hbuf[i] = (_Float16)0.f;
  float cstate[16];
#pragma unroll
  for (int i = 0; i < 16; ++i) cstate[i] = 0.f;
  __syncthreads();

  v8f zero = {0.f, 0.f, 0.f, 0.f, 0.f, 0.f, 0.f, 0.f};

  for (int t = 0; t < TSEQ; ++t) {
    const int teff = dir ? (TSEQ - 1 - t) : t;

    v8f acc[2][4];
#pragma unroll
    for (int mi = 0; mi < 2; ++mi)
#pragma unroll
      for (int g = 0; g < 4; ++g) acc[mi][g] = zero;

    // hWhh for this wave's 32 batch rows x 16 hid cols x 4 gates
    for (int k0 = 0; k0 < HID; k0 += 32) {
      v16h a0 = load_a_frag(hbuf, HID, (mPair * 2 + 0) * 16, k0, lane);
      v16h a1 = load_a_frag(hbuf, HID, (mPair * 2 + 1) * 16, k0, lane);
#pragma unroll
      for (int g = 0; g < 4; ++g) {
        v16h bf = load_b_frag(whh_d, HID, g * HID + hidTile * 16, k0, lane);
        acc[0][g] = wmma_f16(a0, bf, acc[0][g]);
        acc[1][g] = wmma_f16(a1, bf, acc[1][g]);
      }
    }

    float hval[2][8];
#pragma unroll
    for (int mi = 0; mi < 2; ++mi) {
#pragma unroll
      for (int r = 0; r < 8; ++r) {
        int m = (mPair * 2 + mi) * 16 + ((lane >> 4) << 3) + r;  // batch idx
        const float* xrow = xw_d + ((size_t)m * TSEQ + teff) * GATES + hid;
        float pi = acc[mi][0][r] + xrow[0 * HID];
        float pf = acc[mi][1][r] + xrow[1 * HID];
        float pg = acc[mi][2][r] + xrow[2 * HID];
        float po = acc[mi][3][r] + xrow[3 * HID];
        float cn = sigmoidf_(pf) * cstate[mi * 8 + r] + sigmoidf_(pi) * tanhf(pg);
        cstate[mi * 8 + r] = cn;
        float hh = sigmoidf_(po) * tanhf(cn);
        hval[mi][r] = hh;
        out[((size_t)m * TSEQ + teff) * (2 * HID) + dir * HID + hid] = hh;
      }
    }

    __syncthreads();  // all waves done reading old h
#pragma unroll
    for (int mi = 0; mi < 2; ++mi)
#pragma unroll
      for (int r = 0; r < 8; ++r) {
        int m = (mPair * 2 + mi) * 16 + ((lane >> 4) << 3) + r;
        hbuf[m * HID + hid] = (_Float16)hval[mi][r];
      }
    __syncthreads();  // new h visible to all
  }
}

// ---- Kernel: logits = feat @ clsW^T + clsB (N=9, trivial) ------------------
__global__ void classifier_kernel(const float* __restrict__ feat,
                                  const float* __restrict__ Wc,
                                  const float* __restrict__ bc,
                                  float* __restrict__ logits) {
  size_t idx = (size_t)blockIdx.x * blockDim.x + threadIdx.x;
  if (idx >= (size_t)BT * NTAGS) return;
  int tag = (int)(idx % NTAGS);
  size_t token = idx / NTAGS;
  const float* f = feat + token * (2 * HID);
  const float* w = Wc + (size_t)tag * (2 * HID);
  float s = bc[tag];
  for (int k = 0; k < 2 * HID; ++k) s += f[k] * w[k];
  logits[token * NTAGS + tag] = s;
}

// ---- Kernel: CRF NLL, one thread per sequence, block-level reduce ----------
__global__ __launch_bounds__(64) void crf_kernel(
    const float* __restrict__ logits, const int* __restrict__ labels,
    const unsigned char* __restrict__ mask,
    const float* __restrict__ start_trans, const float* __restrict__ end_trans,
    const float* __restrict__ trans, float* __restrict__ loss_out) {
  __shared__ float partial[BBAT];
  __shared__ float tr[NTAGS * NTAGS];
  int tid = threadIdx.x;
  if (tid < NTAGS * NTAGS) tr[tid] = trans[tid];
  __syncthreads();

  int b = tid;
  const float* em = logits + (size_t)b * TSEQ * NTAGS;
  const int* tg = labels + (size_t)b * TSEQ;
  const unsigned char* mk = mask + (size_t)b * TSEQ;

  // numerator
  float num = start_trans[tg[0]] + em[tg[0]];
  int last = 0;
  for (int t = 1; t < TSEQ; ++t) {
    if (mk[t]) {
      num += tr[tg[t - 1] * NTAGS + tg[t]] + em[(size_t)t * NTAGS + tg[t]];
      last = t;
    }
  }
  num += end_trans[tg[last]];

  // forward algorithm (denominator)
  float alpha[NTAGS];
  for (int k = 0; k < NTAGS; ++k) alpha[k] = start_trans[k] + em[k];
  for (int t = 1; t < TSEQ; ++t) {
    if (!mk[t]) continue;
    float na[NTAGS];
    for (int j = 0; j < NTAGS; ++j) {
      float mx = -1e30f;
      for (int i = 0; i < NTAGS; ++i)
        mx = fmaxf(mx, alpha[i] + tr[i * NTAGS + j]);
      float s = 0.f;
      for (int i = 0; i < NTAGS; ++i)
        s += __expf(alpha[i] + tr[i * NTAGS + j] - mx);
      na[j] = mx + __logf(s) + em[(size_t)t * NTAGS + j];
    }
    for (int j = 0; j < NTAGS; ++j) alpha[j] = na[j];
  }
  float mx = -1e30f;
  for (int k = 0; k < NTAGS; ++k) mx = fmaxf(mx, alpha[k] + end_trans[k]);
  float s = 0.f;
  for (int k = 0; k < NTAGS; ++k) s += __expf(alpha[k] + end_trans[k] - mx);
  float denom = mx + __logf(s);

  partial[b] = num - denom;
  __syncthreads();
  if (tid == 0) {
    float tot = 0.f;
    for (int i = 0; i < BBAT; ++i) tot += partial[i];
    loss_out[0] = -tot;
  }
}

// ---------------------------------------------------------------------------
extern "C" void kernel_launch(void* const* d_in, const int* in_sizes, int n_in,
                              void* d_out, int out_size, void* d_ws, size_t ws_size,
                              hipStream_t stream) {
  (void)in_sizes; (void)n_in; (void)out_size; (void)ws_size;

  const int*           input_ids = (const int*)d_in[0];
  const int*           labels    = (const int*)d_in[1];
  const unsigned char* mask      = (const unsigned char*)d_in[2];
  const float*         emb       = (const float*)d_in[3];
  const float* Wih[4]  = {(const float*)d_in[4],  (const float*)d_in[7],
                          (const float*)d_in[10], (const float*)d_in[13]};
  const float* Whh[4]  = {(const float*)d_in[5],  (const float*)d_in[8],
                          (const float*)d_in[11], (const float*)d_in[14]};
  const float* bvec[4] = {(const float*)d_in[6],  (const float*)d_in[9],
                          (const float*)d_in[12], (const float*)d_in[15]};
  const float* clsW   = (const float*)d_in[16];
  const float* clsB   = (const float*)d_in[17];
  const float* startT = (const float*)d_in[18];
  const float* endT   = (const float*)d_in[19];
  const float* trans  = (const float*)d_in[20];
  float* out = (float*)d_out;  // out[0] = loss, out+1 = logits [BT][NTAGS]

  // workspace layout
  char* ws = (char*)d_ws;
  size_t off = 0;
  auto take = [&](size_t bytes) -> void* {
    void* p = (void*)(ws + off);
    off += (bytes + 255) & ~(size_t)255;
    return p;
  };
  _Float16* x16     = (_Float16*)take((size_t)BT * 512 * sizeof(_Float16));
  float*    xw      = (float*)   take((size_t)2 * BT * GATES * sizeof(float));
  float*    feat    = (float*)   take((size_t)BT * 2 * HID * sizeof(float));
  _Float16* wih16_0 = (_Float16*)take((size_t)2 * GATES * EMBP * sizeof(_Float16));
  _Float16* wih16_1 = (_Float16*)take((size_t)2 * GATES * 512 * sizeof(_Float16));
  _Float16* whh16   = (_Float16*)take((size_t)4 * GATES * HID * sizeof(_Float16));

  const int thr = 256;
  auto cvt = [&](const float* s, _Float16* d, int rows, int ks, int kd) {
    size_t total = (size_t)rows * kd;
    unsigned grid = (unsigned)((total + thr - 1) / thr);
    convert_pad_kernel<<<grid, thr, 0, stream>>>(s, d, rows, ks, kd);
  };

  // weight conversion (f32 -> f16, Wih layer0 padded K: 300 -> 320)
  cvt(Wih[0], wih16_0,                        GATES, EMB, EMBP);
  cvt(Wih[1], wih16_0 + (size_t)GATES * EMBP, GATES, EMB, EMBP);
  cvt(Wih[2], wih16_1,                        GATES, 512, 512);
  cvt(Wih[3], wih16_1 + (size_t)GATES * 512,  GATES, 512, 512);
  for (int i = 0; i < 4; ++i)
    cvt(Whh[i], whh16 + (size_t)i * GATES * HID, GATES, HID, HID);

  // embedding gather -> f16 (padded to EMBP)
  embed_kernel<<<BT, EMBP, 0, stream>>>(input_ids, emb, x16);

  dim3 ggrid(BT / 128, GATES / 64);

  // ---- layer 0 ----
  gemm_bias_kernel<<<ggrid, 256, 0, stream>>>(x16, EMBP, wih16_0, EMBP,
                                              bvec[0], xw, GATES, EMBP);
  gemm_bias_kernel<<<ggrid, 256, 0, stream>>>(x16, EMBP,
                                              wih16_0 + (size_t)GATES * EMBP, EMBP,
                                              bvec[1], xw + (size_t)BT * GATES,
                                              GATES, EMBP);
  lstm_scan_kernel<<<2, 1024, 0, stream>>>(xw, whh16, feat);

  // ---- layer 1 ----
  cvt(feat, x16, BT, 512, 512);  // f32 -> f16 activations
  gemm_bias_kernel<<<ggrid, 256, 0, stream>>>(x16, 512, wih16_1, 512,
                                              bvec[2], xw, GATES, 512);
  gemm_bias_kernel<<<ggrid, 256, 0, stream>>>(x16, 512,
                                              wih16_1 + (size_t)GATES * 512, 512,
                                              bvec[3], xw + (size_t)BT * GATES,
                                              GATES, 512);
  lstm_scan_kernel<<<2, 1024, 0, stream>>>(xw, whh16 + (size_t)2 * GATES * HID,
                                           feat);

  // ---- classifier + CRF ----
  {
    size_t total = (size_t)BT * NTAGS;
    unsigned grid = (unsigned)((total + thr - 1) / thr);
    classifier_kernel<<<grid, thr, 0, stream>>>(feat, clsW, clsB, out + 1);
  }
  crf_kernel<<<1, BBAT, 0, stream>>>(out + 1, labels, mask, startT, endT, trans,
                                     out);
}
